// SSMDiagonal_69870527972108
// MI455X (gfx1250) — matrix operand order repaired and checked by reading, not compiled, not run on hardware
//
#include <hip/hip_runtime.h>
#include <hip/hip_bf16.h>

// Problem constants (fixed by reference setup_inputs)
#define BATCH   4
#define LSEQ    4096
#define DDIM    1024
#define NSTATE  64
#define LCHUNK  256
#define NCHUNK  (LSEQ / LCHUNK)     // 16
#define MTOT    (BATCH * LSEQ)      // 16384 tokens

typedef __attribute__((ext_vector_type(16))) __bf16 v16bf;
typedef __attribute__((ext_vector_type(8)))  float  v8f;

// 16 contiguous bf16 (32B) from LDS -> two ds_load_b128
__device__ __forceinline__ v16bf lds_ld_v16(const __bf16* p) {
    v16bf r;
    __builtin_memcpy(&r, p, 32);
    return r;
}
// Two 8-element (16B) chunks at p and p+16 elements (A-fragment pattern)
__device__ __forceinline__ v16bf lds_ld_8_8(const __bf16* p) {
    v16bf r;
    __builtin_memcpy(&r, p, 16);
    __builtin_memcpy(reinterpret_cast<char*>(&r) + 16, p + 16, 16);
    return r;
}

// Async copy 16B global -> LDS (CDNA5, tracked by ASYNCcnt)
__device__ __forceinline__ void async_g2l_b128(void* lds_ptr, const void* gptr) {
    unsigned lo = (unsigned)(uintptr_t)lds_ptr;   // low 32 bits of generic LDS ptr = LDS offset
    asm volatile("global_load_async_to_lds_b128 %0, %1, off"
                 :: "v"(lo), "v"(gptr) : "memory");
}

// ---------------------------------------------------------------------------
// Kernel 0: ZOH discretization parameters (64 threads)
// ---------------------------------------------------------------------------
__global__ void k_params(const float* __restrict__ log_A,
                         const float* __restrict__ B_param,
                         const float* __restrict__ log_dt,
                         float* __restrict__ Abar, float* __restrict__ Bbar,
                         float* __restrict__ Adt,  float* __restrict__ AbarLc) {
    int n = threadIdx.x;
    if (n >= NSTATE) return;
    float dt   = expf(log_dt[0]);
    float A    = -expf(log_A[n]);
    float abar = expf(A * dt);
    bool  sml  = fabsf(A) < 1e-6f;
    float den  = sml ? 1.0f : A;
    float frac = (abar - 1.0f) / den;
    frac       = sml ? dt : frac;
    Abar[n]    = abar;
    Bbar[n]    = frac * B_param[n];
    Adt[n]     = A * dt;
    AbarLc[n]  = expf(A * dt * (float)LCHUNK);
}

// ---------------------------------------------------------------------------
// Kernel 1: u = x @ W_in^T   (M=16384, K=1024, N=64), bf16 WMMA, f32 acc
// Block = 256 threads = 8 waves: 2 row-tiles x 4 col-tiles of 16x16.
// x tile: async global->LDS (f32), double buffered, branch-free pipeline;
//         convert to bf16 at fragment build.
// W_in tile: convert-once bf16 staging, double buffered.
// ---------------------------------------------------------------------------
__global__ __launch_bounds__(256) void k_gemm1(const float* __restrict__ x,
                                               const float* __restrict__ W_in,
                                               float* __restrict__ u) {
    __shared__ float  lds_xf[2][32][36];   // f32 x tile, row stride 144B (16B-aligned chunks)
    __shared__ __bf16 lds_b[2][64][40];    // bf16 W_in tile, row stride 80B

    const int tid  = threadIdx.x;
    const int wave = tid >> 5;
    const int lane = tid & 31;
    const int m0   = blockIdx.x * 32;
    const int rt   = wave >> 2;            // row tile 0..1
    const int ct   = wave & 3;             // col tile 0..3
    const int lr   = lane & 15;
    const int khalf = (lane >> 4) << 3;    // 0 or 8
    const int ks16  = (lane >> 4) << 4;    // 0 or 16

    const int arow = tid >> 3;             // 0..31
    const int akg  = (tid & 7) << 2;       // float idx 0..28 step 4 (16B per thread)
    const int bn   = tid >> 2;             // 0..63
    const int bkg  = (tid & 3) << 3;       // 0,8,16,24

    v8f acc = {0.f, 0.f, 0.f, 0.f, 0.f, 0.f, 0.f, 0.f};

    const float* xrow = x + (size_t)(m0 + arow) * DDIM + akg;

    // Prologue: async-stage x K-slice 0 into buffer 0 (one b128 per thread = full tile)
    async_g2l_b128(&lds_xf[0][arow][akg], xrow);

    int cur = 0;
    for (int kk = 0; kk < DDIM; kk += 32, cur ^= 1) {
        // Stage W_in bf16 tile for this K-slice into bbuf[cur]
        {
            const float* wp = W_in + (size_t)bn * DDIM + kk + bkg;
            const float4 v0 = *reinterpret_cast<const float4*>(wp);
            const float4 v1 = *reinterpret_cast<const float4*>(wp + 4);
            lds_b[cur][bn][bkg + 0] = (__bf16)v0.x;
            lds_b[cur][bn][bkg + 1] = (__bf16)v0.y;
            lds_b[cur][bn][bkg + 2] = (__bf16)v0.z;
            lds_b[cur][bn][bkg + 3] = (__bf16)v0.w;
            lds_b[cur][bn][bkg + 4] = (__bf16)v1.x;
            lds_b[cur][bn][bkg + 5] = (__bf16)v1.y;
            lds_b[cur][bn][bkg + 6] = (__bf16)v1.z;
            lds_b[cur][bn][bkg + 7] = (__bf16)v1.w;
        }
        // Branch-free pipeline: always issue the next-slice async copy into the
        // other buffer. On the last iteration the K offset wraps to 0 -> a
        // harmless in-bounds dummy copy into the dead buffer (never read;
        // drained by S_ENDPGM's implicit wait-idle). Waiting for <=1 in flight
        // guarantees the CURRENT slice (issued last iteration) is complete.
        {
            const int knext = (kk + 32) & (DDIM - 1);
            async_g2l_b128(&lds_xf[cur ^ 1][arow][akg], xrow + knext);
            asm volatile("s_wait_asynccnt 0x1" ::: "memory");
        }
        __syncthreads();

        // A frag: f32 from LDS -> bf16. lane<16: row=lr K{0..7,16..23};
        // lane>=16: K{8..15,24..31}
        v16bf afrag;
        {
            const float* pr = &lds_xf[cur][rt * 16 + lr][0];
            const float4 a0 = *reinterpret_cast<const float4*>(pr + khalf);
            const float4 a1 = *reinterpret_cast<const float4*>(pr + khalf + 4);
            const float4 a2 = *reinterpret_cast<const float4*>(pr + 16 + khalf);
            const float4 a3 = *reinterpret_cast<const float4*>(pr + 16 + khalf + 4);
            afrag[0]  = (__bf16)a0.x; afrag[1]  = (__bf16)a0.y;
            afrag[2]  = (__bf16)a0.z; afrag[3]  = (__bf16)a0.w;
            afrag[4]  = (__bf16)a1.x; afrag[5]  = (__bf16)a1.y;
            afrag[6]  = (__bf16)a1.z; afrag[7]  = (__bf16)a1.w;
            afrag[8]  = (__bf16)a2.x; afrag[9]  = (__bf16)a2.y;
            afrag[10] = (__bf16)a2.z; afrag[11] = (__bf16)a2.w;
            afrag[12] = (__bf16)a3.x; afrag[13] = (__bf16)a3.y;
            afrag[14] = (__bf16)a3.z; afrag[15] = (__bf16)a3.w;
        }
        // B frag: lane<16 col=lr K{0..7,16..23}; lane>=16 K{8..15,24..31}
        v16bf bfrag = lds_ld_8_8(&lds_b[cur][ct * 16 + lr][khalf]);

        acc = __builtin_amdgcn_wmma_f32_16x16x32_bf16(
            false, afrag, false, bfrag, (short)0, acc, false, false);
        __syncthreads();   // all waves done reading buffers before they are reused
    }

    // C/D layout: col = lane&15, row = (lane>>4)*8 + i
    const int orow = m0 + rt * 16 + ((lane >> 4) << 3);
    const int ocol = ct * 16 + lr;
#pragma unroll
    for (int i = 0; i < 8; ++i)
        u[(size_t)(orow + i) * NSTATE + ocol] = acc[i];
}

// ---------------------------------------------------------------------------
// Kernel 2a: per-chunk local scan, in place (h0 = 0), record chunk finals.
// 4096 threads: one per (batch, chunk, n).
// ---------------------------------------------------------------------------
__global__ void k_scan_local(float* __restrict__ u, float* __restrict__ hfinal,
                             const float* __restrict__ Abar,
                             const float* __restrict__ Bbar) {
    int g  = blockIdx.x * blockDim.x + threadIdx.x;   // 0..4095
    int n  = g & (NSTATE - 1);
    int bc = g >> 6;                                  // 0..B*NCHUNK-1
    float a  = Abar[n];
    float bb = Bbar[n];
    float h  = 0.f;
    size_t base = (size_t)bc * LCHUNK * NSTATE + n;
    for (int t = 0; t < LCHUNK; ++t) {
        float v = u[base + (size_t)t * NSTATE];
        h = fmaf(h, a, v * bb);
        u[base + (size_t)t * NSTATE] = h;             // overwrite u with h_local
    }
    hfinal[(size_t)bc * NSTATE + n] = h;
}

// ---------------------------------------------------------------------------
// Kernel 2b: chunk-carry prefix per (b, n): carry_{c+1} = Abar^Lc*carry_c + hfin_c
// ---------------------------------------------------------------------------
__global__ void k_scan_carry(const float* __restrict__ hfinal,
                             float* __restrict__ carr,
                             const float* __restrict__ AbarLc) {
    int tid = threadIdx.x;          // 0..255
    int n   = tid & (NSTATE - 1);
    int b   = tid >> 6;
    float aL = AbarLc[n];
    float carry = 0.f;
    for (int c = 0; c < NCHUNK; ++c) {
        int bc = b * NCHUNK + c;
        carr[(size_t)bc * NSTATE + n] = carry;        // carry entering chunk c
        carry = fmaf(aL, carry, hfinal[(size_t)bc * NSTATE + n]);
    }
}

// ---------------------------------------------------------------------------
// Kernel 3: out = y @ W_out^T with fused fix-up:
//   y[row,n] = (h_local[row,n] + carry[bc,n]*exp(Adt[n]*(t+1))) * C[n]
// Block = 256 threads = 8 waves; tile 16 rows x 128 cols; K = 64 (2 WMMAs).
// ---------------------------------------------------------------------------
__global__ __launch_bounds__(256) void k_gemm2(const float* __restrict__ u,
                                               const float* __restrict__ carr,
                                               const float* __restrict__ Adt,
                                               const float* __restrict__ Cp,
                                               const float* __restrict__ W_out,
                                               float* __restrict__ out) {
    __shared__ __bf16 lds_y[16][72];   // 16 rows x 64 K, stride 72 (144B rows)

    const int tid  = threadIdx.x;
    const int wave = tid >> 5;
    const int lane = tid & 31;
    const int m0   = blockIdx.x * 16;
    const int d0   = blockIdx.y * 128 + wave * 16;

    // Stage fixed-up y tile (16 x 64), 4 elements per thread
    {
        const int r   = tid >> 4;            // 0..15
        const int ng  = (tid & 15) << 2;     // 0..60 step 4
        const int row = m0 + r;
        const int bc  = row / LCHUNK;
        const float tl = (float)(row % LCHUNK + 1);
        const float4 h  = *reinterpret_cast<const float4*>(u    + (size_t)row * NSTATE + ng);
        const float4 cr = *reinterpret_cast<const float4*>(carr + (size_t)bc  * NSTATE + ng);
        const float4 ad = *reinterpret_cast<const float4*>(Adt + ng);
        const float4 cc = *reinterpret_cast<const float4*>(Cp  + ng);
        lds_y[r][ng + 0] = (__bf16)((h.x + cr.x * __expf(ad.x * tl)) * cc.x);
        lds_y[r][ng + 1] = (__bf16)((h.y + cr.y * __expf(ad.y * tl)) * cc.y);
        lds_y[r][ng + 2] = (__bf16)((h.z + cr.z * __expf(ad.z * tl)) * cc.z);
        lds_y[r][ng + 3] = (__bf16)((h.w + cr.w * __expf(ad.w * tl)) * cc.w);
    }
    __syncthreads();

    const int lr    = lane & 15;
    const int khalf = (lane >> 4) << 3;      // 0 or 8
    const int ks16  = (lane >> 4) << 4;      // 0 or 16

    v8f acc = {0.f, 0.f, 0.f, 0.f, 0.f, 0.f, 0.f, 0.f};

#pragma unroll
    for (int kk = 0; kk < NSTATE; kk += 32) {
        v16bf afrag = lds_ld_v16(&lds_y[lr][kk + ks16]);

        // B = W_out (D,N): column d of B-matrix is W_out row d (contiguous K)
        v16bf bfrag;
        const float* wp = W_out + (size_t)(d0 + lr) * NSTATE + kk + ks16;
#pragma unroll
        for (int q = 0; q < 4; ++q) {
            float4 w = reinterpret_cast<const float4*>(wp)[q];
            bfrag[4 * q + 0] = (__bf16)w.x;
            bfrag[4 * q + 1] = (__bf16)w.y;
            bfrag[4 * q + 2] = (__bf16)w.z;
            bfrag[4 * q + 3] = (__bf16)w.w;
        }
        acc = __builtin_amdgcn_wmma_f32_16x16x32_bf16(
            false, afrag, false, bfrag, (short)0, acc, false, false);
    }

    const int orow = m0 + ((lane >> 4) << 3);
    const int ocol = d0 + lr;
#pragma unroll
    for (int i = 0; i < 8; ++i)
        out[(size_t)(orow + i) * DDIM + ocol] = acc[i];
}

// ---------------------------------------------------------------------------
extern "C" void kernel_launch(void* const* d_in, const int* in_sizes, int n_in,
                              void* d_out, int out_size, void* d_ws, size_t ws_size,
                              hipStream_t stream) {
    (void)in_sizes; (void)n_in; (void)out_size; (void)ws_size;
    const float* x      = (const float*)d_in[0];
    const float* W_in   = (const float*)d_in[1];
    const float* W_out  = (const float*)d_in[2];
    const float* log_A  = (const float*)d_in[3];
    const float* B_par  = (const float*)d_in[4];
    const float* C_par  = (const float*)d_in[5];
    const float* log_dt = (const float*)d_in[6];
    float* out = (float*)d_out;

    // Workspace layout (floats): u | hfinal | carr | Abar | Bbar | Adt | AbarLc
    float* ws     = (float*)d_ws;
    float* u      = ws;                                          // 16384*64
    float* hfinal = u + (size_t)MTOT * NSTATE;                   // 4096
    float* carr   = hfinal + (size_t)BATCH * NCHUNK * NSTATE;    // 4096
    float* Abar   = carr + (size_t)BATCH * NCHUNK * NSTATE;
    float* Bbar   = Abar + NSTATE;
    float* Adt    = Bbar + NSTATE;
    float* AbarLc = Adt + NSTATE;

    k_params    <<<1, 64, 0, stream>>>(log_A, B_par, log_dt, Abar, Bbar, Adt, AbarLc);
    k_gemm1     <<<MTOT / 32, 256, 0, stream>>>(x, W_in, u);
    k_scan_local<<<(BATCH * NCHUNK * NSTATE) / 256, 256, 0, stream>>>(u, hfinal, Abar, Bbar);
    k_scan_carry<<<1, 256, 0, stream>>>(hfinal, carr, AbarLc);
    k_gemm2     <<<dim3(MTOT / 16, DDIM / 128), 256, 0, stream>>>(u, carr, Adt, C_par, W_out, out);
}
